// Attention_24764781428921
// MI455X (gfx1250) — matrix operand order repaired and checked by reading, not compiled, not run on hardware
//
#include <hip/hip_runtime.h>
#include <hip/hip_bf16.h>
#include <stdint.h>

#if __has_builtin(__builtin_amdgcn_global_load_async_to_lds_b128)
#define HAVE_ASYNC_LDS 1
#else
#define HAVE_ASYNC_LDS 0
#endif

typedef __attribute__((ext_vector_type(16))) __bf16 v16bf;
typedef __attribute__((ext_vector_type(8)))  __bf16 v8bf;
typedef __attribute__((ext_vector_type(8)))  float  v8f;
typedef __attribute__((ext_vector_type(4)))  int    v4i;

#if HAVE_ASYNC_LDS
typedef __attribute__((address_space(1))) v4i as1_v4i;
typedef __attribute__((address_space(3))) v4i as3_v4i;
#endif

static constexpr int kC   = 1024;
static constexpr int kSeq = 2048;
static constexpr int kB   = 2;
static constexpr int kH   = 16;
static constexpr int kD   = 64;
static constexpr int kTok = kB * kSeq;   // 4096

// ---------------- device helpers ----------------

static __device__ __forceinline__ v8f wmma_bf16(v16bf a, v16bf b, v8f c) {
  // (neg_a, A, neg_b, B, c_mod, C, reuse_a, reuse_b)
  return __builtin_amdgcn_wmma_f32_16x16x32_bf16(false, a, false, b, (short)0, c,
                                                 false, false);
}

// Build a 16-element bf16 fragment from two contiguous 16B chunks.
static __device__ __forceinline__ v16bf ld_frag(const __bf16* base, int o0, int o1) {
  union { v16bf v; v8bf h[2]; } u;
  u.h[0] = *(const v8bf*)(base + o0);
  u.h[1] = *(const v8bf*)(base + o1);
  return u.v;
}

// Copy 32 bytes (16 bf16) global -> LDS, async if the builtin exists.
static __device__ __forceinline__ void stage32(const __bf16* g, __bf16* l) {
#if HAVE_ASYNC_LDS
  __builtin_amdgcn_global_load_async_to_lds_b128((as1_v4i*)g, (as3_v4i*)l, 0, 0);
  __builtin_amdgcn_global_load_async_to_lds_b128((as1_v4i*)(g + 8),
                                                 (as3_v4i*)(l + 8), 0, 0);
#else
  *(v8bf*)l       = *(const v8bf*)g;
  *(v8bf*)(l + 8) = *(const v8bf*)(g + 8);
#endif
}

static __device__ __forceinline__ void stage_wait() {
#if HAVE_ASYNC_LDS
  asm volatile("s_wait_asynccnt 0x0" ::: "memory");
#endif
}

// ---------------- conversion kernels ----------------

__global__ __launch_bounds__(256) void k_cvt_x(const float* __restrict__ x,
                                               __bf16* __restrict__ xh, int n) {
  for (int i = blockIdx.x * blockDim.x + threadIdx.x; i < n;
       i += gridDim.x * blockDim.x)
    xh[i] = (__bf16)x[i];
}

// Wt[n][k] = (bf16) W[k][n], 1024x1024, LDS-tiled transpose.
__global__ __launch_bounds__(256) void k_cvt_wt(const float* __restrict__ W,
                                                __bf16* __restrict__ Wt) {
  __shared__ float t[32][33];
  const int bx = blockIdx.x * 32, by = blockIdx.y * 32;
  const int tx = threadIdx.x & 31, ty = threadIdx.x >> 5;
#pragma unroll
  for (int i = 0; i < 32; i += 8)
    t[ty + i][tx] = W[(size_t)(by + ty + i) * kC + bx + tx];
  __syncthreads();
#pragma unroll
  for (int i = 0; i < 32; i += 8)
    Wt[(size_t)(bx + ty + i) * kC + by + tx] = (__bf16)t[tx][ty + i];
}

// ---------------- GEMM: out = A[M,1024] x Wt^T + bias ----------------
// mode 0: out bf16 [B,H,N,D]     (Q, K)
// mode 1: out bf16 [B,H,D,N]     (V transposed)
// mode 2: out f32  [M,1024]      (final projection)

static constexpr int LDT = 40;   // 32 + 8 bf16 pad (80B row, 16B aligned)

__global__ __launch_bounds__(256) void k_gemm(
    const __bf16* __restrict__ A,    // [M][1024]
    const __bf16* __restrict__ Wt,   // [1024 out][1024 in] (pre-transposed)
    const float*  __restrict__ bias, // [1024]
    void* __restrict__ out, int mode)
{
  __shared__ __bf16 As[2][128 * LDT];
  __shared__ __bf16 Bs[2][128 * LDT];

  const int tid  = threadIdx.x;
  const int lane = tid & 31;
  const int w    = tid >> 5;
  const int hl   = lane & 15;
  const int hi   = lane >> 4;
  const int m0   = blockIdx.y * 128;
  const int n0   = blockIdx.x * 128;
  const int wm   = (w >> 2) * 64;   // 2 wave-rows
  const int wn   = (w & 3) * 32;    // 4 wave-cols

  const int srow = tid >> 1;          // 0..127
  const int scol = (tid & 1) * 16;    // 0 or 16
  const int soff = srow * LDT + scol;

  const __bf16* ga = A  + (size_t)(m0 + srow) * kC + scol;
  const __bf16* gb = Wt + (size_t)(n0 + srow) * kC + scol;

  v8f acc[4][2] = {};

  constexpr int KT = kC / 32;   // 32 K-tiles

  // prologue: tile 0 -> buffer 0
  stage32(ga, &As[0][soff]);
  stage32(gb, &Bs[0][soff]);

  for (int i = 0; i < KT; ++i) {
    const int cur = i & 1;
    stage_wait();          // my async copies for tile i are in LDS
    __syncthreads();       // everyone's tile i visible; buf[!cur] free to reuse

    if (i + 1 < KT) {      // overlap tile i+1 copies with tile i WMMAs
      stage32(ga + (i + 1) * 32, &As[cur ^ 1][soff]);
      stage32(gb + (i + 1) * 32, &Bs[cur ^ 1][soff]);
      if (i + 2 < KT) {
        __builtin_prefetch(ga + (i + 2) * 32, 0, 1);
        __builtin_prefetch(gb + (i + 2) * 32, 0, 1);
      }
    }

    v16bf af[4], bfr[2];
#pragma unroll
    for (int mt = 0; mt < 4; ++mt)
      af[mt] = ld_frag(&As[cur][(wm + mt * 16 + hl) * LDT], hi * 8, 16 + hi * 8);
#pragma unroll
    for (int nt = 0; nt < 2; ++nt)
      bfr[nt] = ld_frag(&Bs[cur][(wn + nt * 16 + hl) * LDT], hi * 8, 16 + hi * 8);
#pragma unroll
    for (int mt = 0; mt < 4; ++mt)
#pragma unroll
      for (int nt = 0; nt < 2; ++nt)
        acc[mt][nt] = wmma_bf16(af[mt], bfr[nt], acc[mt][nt]);
  }

#pragma unroll
  for (int mt = 0; mt < 4; ++mt) {
#pragma unroll
    for (int nt = 0; nt < 2; ++nt) {
#pragma unroll
      for (int r = 0; r < 8; ++r) {
        const int m = m0 + wm + mt * 16 + hi * 8 + r;
        const int n = n0 + wn + nt * 16 + hl;
        const float v = acc[mt][nt][r] + bias[n];
        if (mode == 2) {
          ((float*)out)[(size_t)m * kC + n] = v;
        } else {
          const int b = m >> 11, t = m & (kSeq - 1);
          const int h = n >> 6,  d = n & (kD - 1);
          size_t idx;
          if (mode == 0)
            idx = (((size_t)(b * kH + h) * kSeq) + t) * kD + d;
          else
            idx = (((size_t)(b * kH + h) * kD) + d) * kSeq + t;
          ((__bf16*)out)[idx] = (__bf16)v;
        }
      }
    }
  }
}

// ---------------- flash attention core ----------------

static constexpr int LKV = 72;  // 64 + 8 pad (144B rows, 16B aligned)

__global__ __launch_bounds__(256) void k_attn(
    const __bf16* __restrict__ Q,   // [B*H][Seq][D]
    const __bf16* __restrict__ K,   // [B*H][Seq][D]
    const __bf16* __restrict__ Vt,  // [B*H][D][Seq]
    __bf16* __restrict__ ctx)       // [Tok][C]
{
  __shared__ __bf16 Ks[2][64 * LKV];
  __shared__ __bf16 Vs[2][64 * LKV];
  __shared__ __bf16 Ps[8 * 16 * LKV];

  const int tid = threadIdx.x;
  const int lane = tid & 31, w = tid >> 5;
  const int hl = lane & 15, hi = lane >> 4;
  const int bh = blockIdx.y;          // b*16 + h
  const int qt = blockIdx.x;          // 0..15  (128 query rows per block)

  const __bf16* Qb = Q  + (size_t)bh * kSeq * kD;
  const __bf16* Kb = K  + (size_t)bh * kSeq * kD;
  const __bf16* Vb = Vt + (size_t)bh * kD * kSeq;
  __bf16* Pw = &Ps[w * 16 * LKV];

  // Q fragments: wave owns 16 query rows; A-frag row = hl, K-chunks of 32.
  const int qrow = qt * 128 + w * 16 + hl;
  v16bf qf[2];
#pragma unroll
  for (int kc = 0; kc < 2; ++kc)
    qf[kc] = ld_frag(Qb + (size_t)qrow * kD + kc * 32, hi * 8, 16 + hi * 8);

  const float scale = 0.03125f;   // 1/sqrt(1024)
  float rmax[8], rsum[8];
#pragma unroll
  for (int r = 0; r < 8; ++r) { rmax[r] = -1e30f; rsum[r] = 0.f; }
  v8f o[4] = {};

  const int srow = tid >> 2;          // 0..63
  const int scol = (tid & 3) * 16;    // 0,16,32,48
  const int soff = srow * LKV + scol;

  constexpr int NT = kSeq / 64;       // 32 key tiles

  // prologue: key-tile 0 -> buffer 0
  stage32(Kb + (size_t)srow * kD + scol, &Ks[0][soff]);
  stage32(Vb + (size_t)srow * kSeq + scol, &Vs[0][soff]);

  for (int t = 0; t < NT; ++t) {
    const int cur = t & 1;
    stage_wait();
    __syncthreads();

    if (t + 1 < NT) {   // overlap next K/V tile copy with this tile's math
      const int kt = (t + 1) * 64;
      stage32(Kb + (size_t)(kt + srow) * kD + scol, &Ks[cur ^ 1][soff]);
      stage32(Vb + (size_t)srow * kSeq + kt + scol, &Vs[cur ^ 1][soff]);
    }

    // S = Q x K^T  (16x64 tile): 4 key-tiles x 2 d-chunks of WMMA
    v8f s[4] = {};
#pragma unroll
    for (int nt = 0; nt < 4; ++nt) {
      const __bf16* kb = &Ks[cur][(nt * 16 + hl) * LKV];
#pragma unroll
      for (int kc = 0; kc < 2; ++kc)
        s[nt] = wmma_bf16(qf[kc],
                          ld_frag(kb, kc * 32 + hi * 8, kc * 32 + 16 + hi * 8),
                          s[nt]);
    }

    // tile row-max (row m = r + hi*8; keys striped over 16 lanes + nt)
    float tmax[8];
#pragma unroll
    for (int r = 0; r < 8; ++r) {
      float z = fmaxf(fmaxf(s[0][r], s[1][r]), fmaxf(s[2][r], s[3][r]));
      tmax[r] = z * scale;
    }
#pragma unroll
    for (int msk = 1; msk <= 8; msk <<= 1)
#pragma unroll
      for (int r = 0; r < 8; ++r)
        tmax[r] = fmaxf(tmax[r], __shfl_xor(tmax[r], msk, 32));

    float nm[8], corr[8], ts[8];
#pragma unroll
    for (int r = 0; r < 8; ++r) {
      nm[r]   = fmaxf(rmax[r], tmax[r]);
      corr[r] = __expf(rmax[r] - nm[r]);
      ts[r]   = 0.f;
    }
    // probabilities -> per-wave LDS (reshape into A-fragment layout)
#pragma unroll
    for (int nt = 0; nt < 4; ++nt)
#pragma unroll
      for (int r = 0; r < 8; ++r) {
        const float p = __expf(s[nt][r] * scale - nm[r]);
        ts[r] += p;
        Pw[(r + hi * 8) * LKV + nt * 16 + hl] = (__bf16)p;
      }
#pragma unroll
    for (int msk = 1; msk <= 8; msk <<= 1)
#pragma unroll
      for (int r = 0; r < 8; ++r)
        ts[r] += __shfl_xor(ts[r], msk, 32);
#pragma unroll
    for (int r = 0; r < 8; ++r) {
      rsum[r] = rsum[r] * corr[r] + ts[r];
      rmax[r] = nm[r];
    }
#pragma unroll
    for (int dt = 0; dt < 4; ++dt)
#pragma unroll
      for (int r = 0; r < 8; ++r)
        o[dt][r] = o[dt][r] * corr[r];

    // ctx += P x V : A from Pw, B from Vs[d][key]
#pragma unroll
    for (int kc = 0; kc < 2; ++kc) {
      const v16bf pf = ld_frag(&Pw[hl * LKV], kc * 32 + hi * 8, kc * 32 + 16 + hi * 8);
#pragma unroll
      for (int dt = 0; dt < 4; ++dt) {
        const v16bf vf = ld_frag(&Vs[cur][(dt * 16 + hl) * LKV],
                                 kc * 32 + hi * 8, kc * 32 + 16 + hi * 8);
        o[dt] = wmma_bf16(pf, vf, o[dt]);
      }
    }
  }

  // normalize + write ctx as token-major bf16 [Tok][C]
  const int b = bh >> 4, h = bh & 15;
#pragma unroll
  for (int dt = 0; dt < 4; ++dt)
#pragma unroll
    for (int r = 0; r < 8; ++r) {
      const int qn = qt * 128 + w * 16 + r + hi * 8;
      const int c  = h * kD + dt * 16 + hl;
      ctx[(size_t)(b * kSeq + qn) * kC + c] = (__bf16)(o[dt][r] / rsum[r]);
    }
}

// ---------------- host launch ----------------

extern "C" void kernel_launch(void* const* d_in, const int* in_sizes, int n_in,
                              void* d_out, int out_size, void* d_ws, size_t ws_size,
                              hipStream_t stream) {
  (void)in_sizes; (void)n_in; (void)out_size; (void)ws_size;
  const float* x  = (const float*)d_in[0];
  const float* Wq = (const float*)d_in[1];
  const float* bq = (const float*)d_in[2];
  const float* Wk = (const float*)d_in[3];
  const float* bk = (const float*)d_in[4];
  const float* Wv = (const float*)d_in[5];
  const float* bv = (const float*)d_in[6];
  const float* Wo = (const float*)d_in[7];
  const float* bo = (const float*)d_in[8];

  char* ws = (char*)d_ws;
  const size_t MB = 1ull << 20;
  __bf16* xh  = (__bf16*)(ws + 0);        // 8 MB  x in bf16
  __bf16* Wqt = (__bf16*)(ws + 8  * MB);  // 2 MB each, transposed bf16
  __bf16* Wkt = (__bf16*)(ws + 10 * MB);
  __bf16* Wvt = (__bf16*)(ws + 12 * MB);
  __bf16* Wot = (__bf16*)(ws + 14 * MB);
  __bf16* Qh  = (__bf16*)(ws + 16 * MB);  // 8 MB [B,H,N,D]
  __bf16* Kh  = (__bf16*)(ws + 24 * MB);  // 8 MB [B,H,N,D]
  __bf16* Vt  = (__bf16*)(ws + 32 * MB);  // 8 MB [B,H,D,N]
  __bf16* Ctx = (__bf16*)(ws + 40 * MB);  // 8 MB [Tok,C]

  k_cvt_x<<<1024, 256, 0, stream>>>(x, xh, kTok * kC);
  dim3 tg(32, 32);
  k_cvt_wt<<<tg, 256, 0, stream>>>(Wq, Wqt);
  k_cvt_wt<<<tg, 256, 0, stream>>>(Wk, Wkt);
  k_cvt_wt<<<tg, 256, 0, stream>>>(Wv, Wvt);
  k_cvt_wt<<<tg, 256, 0, stream>>>(Wo, Wot);

  dim3 gg(kC / 128, kTok / 128);   // (8, 32)
  k_gemm<<<gg, 256, 0, stream>>>(xh, Wqt, bq, Qh, 0);
  k_gemm<<<gg, 256, 0, stream>>>(xh, Wkt, bk, Kh, 0);
  k_gemm<<<gg, 256, 0, stream>>>(xh, Wvt, bv, Vt, 1);

  k_attn<<<dim3(kSeq / 128, kB * kH), 256, 0, stream>>>(Qh, Kh, Vt, Ctx);

  k_gemm<<<gg, 256, 0, stream>>>(Ctx, Wot, bo, d_out, 2);
}
